// EdgeDecoder_32186484916768
// MI455X (gfx1250) — compile-verified, matrix operand order checked by reading, not run on hardware
//
#include <hip/hip_runtime.h>
#include <hip/hip_bf16.h>

// ---------------------------------------------------------------------------
// EdgeDecoder on MI455X (gfx1250):
//   out[e] = W2 . relu(W1 . concat(zm[row[e]], zd[col[e]]) + b1) + b2
// bf16 WMMA GEMM, W1 resident in LDS (256 KB), 32-edge M-tiles, persistent
// blocks. Optional pre-converted bf16 embedding tables in workspace with
// async global->LDS gather (ASYNCcnt) when available.
// ---------------------------------------------------------------------------

#define HIDDEN   256
#define KDIM     512            // 2*HIDDEN concat width
#define TILE_M   32             // edges per tile (2 WMMA M-subtiles)
#define NTHREADS 256            // 8 waves (wave32)

typedef __attribute__((ext_vector_type(16))) __bf16 v16bf;
typedef __attribute__((ext_vector_type(8)))  float  v8f;
typedef __attribute__((ext_vector_type(4)))  int    v4i;

#if defined(__has_builtin)
#if __has_builtin(__builtin_amdgcn_global_load_async_to_lds_b128)
#define HAVE_ASYNC_B128 1
#endif
#if __has_builtin(__builtin_amdgcn_s_wait_asynccnt)
#define HAVE_WAIT_ASYNC 1
#endif
#endif

union Frag32B {                 // 32-byte fragment: 16 bf16 == two uint4
    v16bf bf;
    uint4 u[2];
};

__device__ __forceinline__ unsigned int bf16_bits(float f) {
    unsigned int u = __float_as_uint(f);
    unsigned int r = u + 0x7FFFu + ((u >> 16) & 1u);   // round-to-nearest-even
    return r >> 16;
}
__device__ __forceinline__ unsigned int pack2bf(float lo, float hi) {
    return (bf16_bits(hi) << 16) | bf16_bits(lo);
}

// ---- prep: f32 -> packed bf16 pairs ---------------------------------------
__global__ void cvt_f32_bf16(const float* __restrict__ src,
                             unsigned int* __restrict__ dst_u32, int n_pairs) {
    int i = blockIdx.x * blockDim.x + threadIdx.x;   // one uint = 2 bf16
    if (i < n_pairs) {
        float2 f = ((const float2*)src)[i];
        dst_u32[i] = pack2bf(f.x, f.y);
    }
}

// ---- main fused kernel -----------------------------------------------------
// BFTAB = true: zmp/zdp point at bf16 tables (pure copy gather, async path).
// BFTAB = false: zmp/zdp point at the original f32 tables (convert on gather).
template <bool BFTAB>
__global__ __launch_bounds__(NTHREADS, 1)
void edge_mlp_wmma(const void* __restrict__ zmp,
                   const void* __restrict__ zdp,
                   const long long* __restrict__ eidx,      // [2][E] int64
                   const unsigned short* __restrict__ w1bf, // bf16 [256][512]
                   const float* __restrict__ b1,
                   const float* __restrict__ w2,
                   const float* __restrict__ b2,
                   float* __restrict__ out,
                   int nE) {
    // LDS: 262144 + 32768 + 128 = 295040 B (<= 320 KB WGP LDS, 1 WG/WGP)
    __shared__ __align__(16) unsigned short sW1[HIDDEN * KDIM]; // bf16 bits
    __shared__ __align__(16) unsigned short sA [TILE_M * KDIM]; // bf16 bits
    __shared__ float sOut[TILE_M];

    const int tid   = threadIdx.x;
    const int lane  = tid & 31;
    const int wave  = tid >> 5;         // 0..7 -> N slab [wave*32, wave*32+32)
    const int n0    = wave * 32;
    const int hlane = lane & 15;        // position within 16-lane half
    const int koff  = (lane < 16) ? 0 : 8;

    // ---- stage W1 (bf16) into LDS once per block --------------------------
    {
        const uint4* src = (const uint4*)w1bf;
        uint4*       dst = (uint4*)sW1;
        const int n16 = HIDDEN * KDIM * 2 / 16;     // 16384 uint4
        for (int i = tid; i < n16; i += NTHREADS) dst[i] = src[i];
    }

    const int ntiles = (nE + TILE_M - 1) / TILE_M;  // 31250 for E=1e6

    for (int t = blockIdx.x; t < ntiles; t += gridDim.x) {
        if (tid < TILE_M) sOut[tid] = 0.0f;

        // ---- gather the 32x512 A tile (bf16) into LDS ---------------------
        {
            const int m   = tid >> 3;        // edge within tile (0..31)
            const int seg = tid & 7;         // 64-column segment of concat row
            long long e = (long long)t * TILE_M + m;
            if (e >= nE) e = nE - 1;         // clamp (output store is guarded)
            const long long r = eidx[e];                 // -> microbes
            const long long c = eidx[(long long)nE + e]; // -> diseases

            if (BFTAB) {
                // pure 128-byte copy per thread from bf16 tables
                const unsigned short* zm = (const unsigned short*)zmp;
                const unsigned short* zd = (const unsigned short*)zdp;
                const uint4* s4 = (seg < 4)
                    ? (const uint4*)(zm + r * HIDDEN + seg * 64)
                    : (const uint4*)(zd + c * HIDDEN + (seg - 4) * 64);
                uint4* d4 = (uint4*)&sA[m * KDIM + seg * 64];
#ifdef HAVE_ASYNC_B128
                #pragma unroll
                for (int i = 0; i < 8; ++i)
                    __builtin_amdgcn_global_load_async_to_lds_b128(
                        (__attribute__((address_space(1))) v4i*)(s4 + i),
                        (__attribute__((address_space(3))) v4i*)(d4 + i),
                        0, 0);
#ifdef HAVE_WAIT_ASYNC
                __builtin_amdgcn_s_wait_asynccnt(0);
#else
                asm volatile("s_wait_asynccnt 0x0" ::: "memory");
#endif
#else
                #pragma unroll
                for (int i = 0; i < 8; ++i) d4[i] = s4[i];
#endif
            } else {
                // f32 gather + convert: 64 floats per thread
                const float* zm = (const float*)zmp;
                const float* zd = (const float*)zdp;
                const float4* s4 = (seg < 4)
                    ? (const float4*)(zm + r * HIDDEN + seg * 64)
                    : (const float4*)(zd + c * HIDDEN + (seg - 4) * 64);
                uint4* d4 = (uint4*)&sA[m * KDIM + seg * 64];
                #pragma unroll
                for (int i = 0; i < 8; ++i) {            // 8 floats / iter
                    float4 a = s4[2 * i + 0];
                    float4 b = s4[2 * i + 1];
                    uint4 p;
                    p.x = pack2bf(a.x, a.y);
                    p.y = pack2bf(a.z, a.w);
                    p.z = pack2bf(b.x, b.y);
                    p.w = pack2bf(b.z, b.w);
                    d4[i] = p;
                }
            }
        }
        __syncthreads();

        // ---- layer 1: [32 x 512] x [512 x 32-slab] via bf16 WMMA ----------
        // acc[s][j]: M-subtile s (rows s*16..s*16+15), N-tile j. Bias init.
        v8f acc[2][2];
        {
            float bv0 = b1[n0 + hlane];
            float bv1 = b1[n0 + 16 + hlane];
            #pragma unroll
            for (int v = 0; v < 8; ++v) {
                acc[0][0][v] = bv0; acc[0][1][v] = bv1;
                acc[1][0][v] = bv0; acc[1][1][v] = bv1;
            }
        }

        const int nrow0 = n0 + hlane;        // W1 row feeding B-frag ntile 0
        const int nrow1 = n0 + 16 + hlane;   // ntile 1

        #pragma unroll 2
        for (int kb = 0; kb < KDIM; kb += 32) {
            // B fragments (shared by both M-subtiles): contiguous W1 rows
            Frag32B fb0, fb1;
            {
                const unsigned short* bp0 = &sW1[nrow0 * KDIM + kb + koff];
                fb0.u[0] = *(const uint4*)(bp0);
                fb0.u[1] = *(const uint4*)(bp0 + 16);
                const unsigned short* bp1 = &sW1[nrow1 * KDIM + kb + koff];
                fb1.u[0] = *(const uint4*)(bp1);
                fb1.u[1] = *(const uint4*)(bp1 + 16);
            }
            #pragma unroll
            for (int s = 0; s < 2; ++s) {
                Frag32B fa;
                const unsigned short* ap =
                    &sA[(s * 16 + hlane) * KDIM + kb + koff];
                fa.u[0] = *(const uint4*)(ap);
                fa.u[1] = *(const uint4*)(ap + 16);
                acc[s][0] = __builtin_amdgcn_wmma_f32_16x16x32_bf16(
                    false, fa.bf, false, fb0.bf, (short)0, acc[s][0],
                    false, false);
                acc[s][1] = __builtin_amdgcn_wmma_f32_16x16x32_bf16(
                    false, fa.bf, false, fb1.bf, (short)0, acc[s][1],
                    false, false);
            }
        }

        // ---- layer 2: out[m] = b2 + sum_n relu(h[m][n]) * W2[n] -----------
        // C layout: VGPR v, lane l -> M = v + (l<16?0:8), N = slab + (l&15).
        float w20 = w2[n0 + hlane];
        float w21 = w2[n0 + 16 + hlane];
        #pragma unroll
        for (int s = 0; s < 2; ++s) {
            float p[8];
            #pragma unroll
            for (int v = 0; v < 8; ++v)
                p[v] = fmaxf(acc[s][0][v], 0.0f) * w20 +
                       fmaxf(acc[s][1][v], 0.0f) * w21;
            // xor masks 1,2,4,8 stay within each 16-lane half
            #pragma unroll
            for (int off = 1; off < 16; off <<= 1) {
                #pragma unroll
                for (int v = 0; v < 8; ++v)
                    p[v] += __shfl_xor(p[v], off, 32);
            }
            if (hlane == 0) {
                const int mbase = s * 16 + ((lane < 16) ? 0 : 8);
                #pragma unroll
                for (int v = 0; v < 8; ++v)
                    atomicAdd(&sOut[mbase + v], p[v]);   // ds_add_f32
            }
        }
        __syncthreads();

        if (tid < TILE_M) {
            long long e = (long long)t * TILE_M + tid;
            if (e < nE) out[e] = sOut[tid] + b2[0];
        }
        // Next trip's sA/sOut writes are ordered by its pre-compute barrier;
        // this trip's compute finished before the barrier above.
    }
}

// ---------------------------------------------------------------------------
extern "C" void kernel_launch(void* const* d_in, const int* in_sizes, int n_in,
                              void* d_out, int out_size, void* d_ws, size_t ws_size,
                              hipStream_t stream) {
    const float*      zm   = (const float*)d_in[0];     // [50000,256]
    const float*      zd   = (const float*)d_in[1];     // [50000,256]
    const long long*  eidx = (const long long*)d_in[2]; // [2,E] int64
    const float*      W1   = (const float*)d_in[3];     // [256,512]
    const float*      b1   = (const float*)d_in[4];     // [256]
    const float*      W2   = (const float*)d_in[5];     // [1,256]
    const float*      b2   = (const float*)d_in[6];     // [1]
    float*            out  = (float*)d_out;

    const int nE  = in_sizes[2] / 2;
    const int nzm = in_sizes[0];                        // 12.8M floats
    const int nzd = in_sizes[1];

    // workspace layout: [0,256K) W1 bf16 ; then zm bf16 ; then zd bf16
    const size_t w1Bytes  = (size_t)HIDDEN * KDIM * 2;  // 262144
    const size_t tabBytes = ((size_t)nzm + nzd) * 2;    // 51.2 MB
    unsigned char* ws = (unsigned char*)d_ws;
    unsigned int* w1bf_u32 = (unsigned int*)ws;

    const int w1_pairs = HIDDEN * KDIM / 2;             // 65536 uints
    cvt_f32_bf16<<<(w1_pairs + 255) / 256, 256, 0, stream>>>(W1, w1bf_u32, w1_pairs);

    const bool big_ws = ws_size >= w1Bytes + tabBytes;

    const int ntiles = (nE + TILE_M - 1) / TILE_M;
    int blocks = ntiles < 2048 ? ntiles : 2048;         // persistent blocks

    if (big_ws) {
        unsigned int* zmb = (unsigned int*)(ws + w1Bytes);
        unsigned int* zdb = (unsigned int*)(ws + w1Bytes + (size_t)nzm * 2);
        cvt_f32_bf16<<<(nzm / 2 + 255) / 256, 256, 0, stream>>>(zm, zmb, nzm / 2);
        cvt_f32_bf16<<<(nzd / 2 + 255) / 256, 256, 0, stream>>>(zd, zdb, nzd / 2);
        edge_mlp_wmma<true><<<blocks, NTHREADS, 0, stream>>>(
            (const void*)zmb, (const void*)zdb, eidx,
            (const unsigned short*)ws, b1, W2, b2, out, nE);
    } else {
        edge_mlp_wmma<false><<<blocks, NTHREADS, 0, stream>>>(
            (const void*)zm, (const void*)zd, eidx,
            (const unsigned short*)ws, b1, W2, b2, out, nE);
    }
}
